// GraphHopfieldLayer_41832981463305
// MI455X (gfx1250) — matrix operand order repaired and checked by reading, not compiled, not run on hardware
//
#include <hip/hip_runtime.h>
#include <hip/hip_bf16.h>

#define DDIM   256
#define KMEM   1024
#define TILE_M 16
#define WPB    4          // waves (16-row tiles) per block
#define BETA_C   1.0f
#define ALPHA_C  0.5f
#define LAPC     0.2f     // 2 * LAMBDA_GRAPH
#define LN_EPS_C 1e-5f

typedef __attribute__((ext_vector_type(16))) __bf16 v16bf;
typedef __attribute__((ext_vector_type(8)))  __bf16 v8bf;
typedef __attribute__((ext_vector_type(8)))  float  v8f;

// ---- one-time: memory patterns -> bf16, row-major (Mb) and transposed (Mt) ----
__global__ void ghl_prep_mem(const float* __restrict__ M, __bf16* __restrict__ Mb,
                             __bf16* __restrict__ Mt) {
  int i = blockIdx.x * blockDim.x + threadIdx.x;   // [0, K*D), exact grid
  int k = i >> 8, d = i & 255;
  float v = M[i];
  Mb[i] = (__bf16)v;
  Mt[(size_t)d * KMEM + k] = (__bf16)v;
}

// ---- degree count (scatter over col), then deg -> deg^{-1/2} in place ----
__global__ void ghl_deg(const int* __restrict__ col, float* __restrict__ deg, int E) {
  int e = blockIdx.x * blockDim.x + threadIdx.x;
  if (e < E) atomicAdd(&deg[col[e]], 1.0f);
}
__global__ void ghl_dinv(float* __restrict__ deg, int N) {
  int i = blockIdx.x * blockDim.x + threadIdx.x;
  if (i < N) {
    float d = deg[i];
    deg[i] = (d > 0.0f) ? rsqrtf(fmaxf(d, 1.0f)) : 0.0f;
  }
}

// ---- x (fp32) -> xb (bf16), 8 elements per thread ----
__global__ void ghl_f2bf(const float* __restrict__ in, __bf16* __restrict__ out, long long n8) {
  long long i = blockIdx.x * (long long)blockDim.x + threadIdx.x;
  if (i >= n8) return;
  const float4* p = (const float4*)(in + i * 8);
  float4 a = p[0], b = p[1];
  v8bf o;
  o[0] = (__bf16)a.x; o[1] = (__bf16)a.y; o[2] = (__bf16)a.z; o[3] = (__bf16)a.w;
  o[4] = (__bf16)b.x; o[5] = (__bf16)b.y; o[6] = (__bf16)b.z; o[7] = (__bf16)b.w;
  *(v8bf*)(out + i * 8) = o;
}

// ---- edge scatter: agg[row] += deg_inv_sqrt[col] * x[col], 8 floats/thread ----
__global__ void ghl_scatter(const int* __restrict__ rowi, const int* __restrict__ coli,
                            const float* __restrict__ x, const float* __restrict__ dis,
                            float* __restrict__ agg, long long nwork) {
  long long idx = blockIdx.x * (long long)blockDim.x + threadIdx.x;
  if (idx >= nwork) return;
  int e = (int)(idx >> 5);
  int c = ((int)idx & 31) * 8;
  int r = rowi[e], cl = coli[e];
  float sc = dis[cl];
  const float4* xs = (const float4*)(x + (size_t)cl * DDIM + c);
  float4 a = xs[0], b = xs[1];
  float* ap = agg + (size_t)r * DDIM + c;
  atomicAdd(ap + 0, sc * a.x); atomicAdd(ap + 1, sc * a.y);
  atomicAdd(ap + 2, sc * a.z); atomicAdd(ap + 3, sc * a.w);
  atomicAdd(ap + 4, sc * b.x); atomicAdd(ap + 5, sc * b.y);
  atomicAdd(ap + 6, sc * b.z); atomicAdd(ap + 7, sc * b.w);
}

// ---- fused Hopfield attention (flash-style over K=1024) + combine + LayerNorm ----
// One wave per 16-node tile. Per 32-memory-row chunk:
//   S[16,32]  = X[16,256] @ M_chunk^T   (8 K-steps x 2 WMMA, fp32 acc)
//   online softmax (row reduce across 16 lanes), P -> LDS (bf16) -> A-fragment
//   O[16,256] += P[16,32] @ M_chunk     (16 WMMA)
// Epilogue: x_new = (1-a)x + a*(O/l - LAPC*(x - dis*agg)); LayerNorm; store.
__global__ __launch_bounds__(WPB * 32)
void ghl_hopfield(const __bf16* __restrict__ xb, const __bf16* __restrict__ Mb,
                  const __bf16* __restrict__ Mt, const float* __restrict__ xcur,
                  const float* __restrict__ agg, const float* __restrict__ dis,
                  const float* __restrict__ gma, const float* __restrict__ bta,
                  float* __restrict__ xout, int nrows, int ntiles) {
  __shared__ __align__(16) __bf16 lx[WPB][TILE_M][DDIM];  // 4*8KB staged X tiles
  __shared__ __align__(16) __bf16 lp[WPB][TILE_M][32];    // wave-private P relayout

  const int lane = threadIdx.x & 31;
  const int wv   = threadIdx.x >> 5;
  const int tile = blockIdx.x * WPB + wv;
  if (tile >= ntiles) return;                // no barriers used -> divergent exit ok
  const int row0 = tile * TILE_M;
  const int mrow = lane & 15;
  const int half = lane >> 4;

  // ---- async-stage X tile (16x256 bf16 = 8KB contiguous) into LDS ----
  // GLOBAL_LOAD_ASYNC_TO_LDS_B128: memory -> LDS direct, tracked with ASYNCcnt
  // (GV addressing: VDST = per-lane LDS byte offset, VADDR = 64-bit global addr)
  {
    unsigned lbase = (unsigned)(size_t)&lx[wv][0][0];
    unsigned long long gbase = (unsigned long long)(xb + (size_t)row0 * DDIM);
    #pragma unroll
    for (int i = 0; i < 16; ++i) {
      unsigned           loff = lbase + (unsigned)(lane * 16 + i * 512);
      unsigned long long gadr = gbase + (unsigned long long)(lane * 16 + i * 512);
      asm volatile("global_load_async_to_lds_b128 %0, %1, off"
                   :: "v"(loff), "v"(gadr) : "memory");
    }
    asm volatile("s_wait_asynccnt 0x0" ::: "memory");
  }

  float mrun[8], lrun[8];
  #pragma unroll
  for (int j = 0; j < 8; ++j) { mrun[j] = -3.0e38f; lrun[j] = 0.0f; }
  v8f O[16] = {};                            // 16x256 fp32 accumulator (128 VGPRs)

  const __bf16* xr = &lx[wv][mrow][0];

  #pragma unroll 1
  for (int cb = 0; cb < KMEM; cb += 32) {
    // ---------- GEMM1: scores ----------
    v8f c0 = {}, c1 = {};
    const __bf16* b0p = Mb + (size_t)(cb + mrow) * DDIM + half * 16;
    const __bf16* b1p = Mb + (size_t)(cb + 16 + mrow) * DDIM + half * 16;
    #pragma unroll
    for (int dc = 0; dc < DDIM; dc += 32) {
      v8bf alo = *(const v8bf*)(xr + dc + half * 16);
      v8bf ahi = *(const v8bf*)(xr + dc + half * 16 + 8);
      v16bf a = __builtin_shufflevector(alo, ahi, 0,1,2,3,4,5,6,7,8,9,10,11,12,13,14,15);
      v16bf b0 = *(const v16bf*)(b0p + dc);
      v16bf b1 = *(const v16bf*)(b1p + dc);
      c0 = __builtin_amdgcn_wmma_f32_16x16x32_bf16(false, a, false, b0, (short)0, c0, false, false);
      c1 = __builtin_amdgcn_wmma_f32_16x16x32_bf16(false, a, false, b1, (short)0, c1, false, false);
    }
    // ---------- online softmax over 32 new columns ----------
    float osc[8];
    #pragma unroll
    for (int j = 0; j < 8; ++j) {
      float s0 = BETA_C * c0[j];
      float s1 = BETA_C * c1[j];
      float t = fmaxf(s0, s1);
      #pragma unroll
      for (int off = 1; off < 16; off <<= 1) t = fmaxf(t, __shfl_xor(t, off, 16));
      float mnew = fmaxf(mrun[j], t);
      osc[j] = __expf(mrun[j] - mnew);
      mrun[j] = mnew;
      float p0 = __expf(s0 - mnew);
      float p1 = __expf(s1 - mnew);
      float rs = p0 + p1;
      #pragma unroll
      for (int off = 1; off < 16; off <<= 1) rs += __shfl_xor(rs, off, 16);
      lrun[j] = lrun[j] * osc[j] + rs;
      lp[wv][j + 8 * half][mrow]      = (__bf16)p0;   // C-layout -> A-layout via LDS
      lp[wv][j + 8 * half][16 + mrow] = (__bf16)p1;
    }
    // rescale running output accumulator
    #pragma unroll
    for (int t = 0; t < 16; ++t) {
      #pragma unroll
      for (int j = 0; j < 8; ++j) O[t][j] *= osc[j];
    }
    // ---------- GEMM2: O += P @ M_chunk ----------
    v8bf plo = *(const v8bf*)&lp[wv][mrow][half * 16];
    v8bf phi = *(const v8bf*)&lp[wv][mrow][half * 16 + 8];
    v16bf ap = __builtin_shufflevector(plo, phi, 0,1,2,3,4,5,6,7,8,9,10,11,12,13,14,15);
    #pragma unroll
    for (int t = 0; t < 16; ++t) {
      v16bf bm = *(const v16bf*)(Mt + (size_t)(t * 16 + mrow) * KMEM + cb + half * 16);
      O[t] = __builtin_amdgcn_wmma_f32_16x16x32_bf16(false, ap, false, bm, (short)0, O[t], false, false);
    }
  }

  // ---------- epilogue: combine with Laplacian, LayerNorm, store ----------
  #pragma unroll
  for (int j = 0; j < 8; ++j) {
    int r = row0 + j + 8 * half;
    int rr = (r < nrows) ? r : 0;
    float invl = 1.0f / lrun[j];
    float di = dis[rr];
    float musum = 0.0f;
    float xn[16];
    #pragma unroll
    for (int t = 0; t < 16; ++t) {
      int d = t * 16 + mrow;
      float xv  = xcur[(size_t)rr * DDIM + d];
      float lap = xv - di * agg[(size_t)rr * DDIM + d];
      float v = (1.0f - ALPHA_C) * xv + ALPHA_C * (O[t][j] * invl - LAPC * lap);
      xn[t] = v;
      musum += v;
    }
    #pragma unroll
    for (int off = 1; off < 16; off <<= 1) musum += __shfl_xor(musum, off, 16);
    float mu = musum * (1.0f / DDIM);
    float vs = 0.0f;
    #pragma unroll
    for (int t = 0; t < 16; ++t) { float dv = xn[t] - mu; vs += dv * dv; }
    #pragma unroll
    for (int off = 1; off < 16; off <<= 1) vs += __shfl_xor(vs, off, 16);
    float rstd = rsqrtf(vs * (1.0f / DDIM) + LN_EPS_C);
    if (r < nrows) {
      #pragma unroll
      for (int t = 0; t < 16; ++t) {
        int d = t * 16 + mrow;
        xout[(size_t)r * DDIM + d] = (xn[t] - mu) * rstd * gma[d] + bta[d];
      }
    }
  }
}

extern "C" void kernel_launch(void* const* d_in, const int* in_sizes, int n_in,
                              void* d_out, int out_size, void* d_ws, size_t ws_size,
                              hipStream_t stream) {
  (void)n_in; (void)out_size; (void)ws_size;
  const float* x    = (const float*)d_in[0];
  const int*   eidx = (const int*)d_in[1];   // edge_index [2, E] (int per harness)
  const float* mem  = (const float*)d_in[2];
  const float* gma  = (const float*)d_in[3];
  const float* bta  = (const float*)d_in[4];
  float* out = (float*)d_out;

  const int E = in_sizes[1] / 2;
  const int N = in_sizes[0] / DDIM;
  const int* rowi = eidx;
  const int* coli = eidx + E;

  const int ntiles    = (N + TILE_M - 1) / TILE_M;
  const int nrows_pad = ntiles * TILE_M;

  // workspace carve (512B aligned slices)
  char* ws = (char*)d_ws;
  auto take = [&](size_t bytes) { char* p = ws; ws += (bytes + 511) & ~(size_t)511; return p; };
  __bf16* Mb  = (__bf16*)take((size_t)KMEM * DDIM * 2);        // 512 KB
  __bf16* Mt  = (__bf16*)take((size_t)DDIM * KMEM * 2);        // 512 KB
  float*  dis = (float*) take((size_t)N * 4);                  // 400 KB
  __bf16* xb  = (__bf16*)take((size_t)nrows_pad * DDIM * 2);   // ~51 MB
  float*  agg = (float*) take((size_t)N * DDIM * 4);           // ~102 MB

  // memory patterns -> bf16 (row-major + transposed); stays L2 resident (1 MB)
  ghl_prep_mem<<<(KMEM * DDIM) / 256, 256, 0, stream>>>(mem, Mb, Mt);

  // degree -> deg^{-1/2} once (loop-invariant in the reference)
  hipMemsetAsync(dis, 0, (size_t)N * 4, stream);
  ghl_deg<<<(E + 255) / 256, 256, 0, stream>>>(coli, dis, E);
  ghl_dinv<<<(N + 255) / 256, 256, 0, stream>>>(dis, N);

  if (nrows_pad > N)
    hipMemsetAsync(xb + (size_t)N * DDIM, 0, (size_t)(nrows_pad - N) * DDIM * 2, stream);

  const long long scat_work = (long long)E * 32;
  const long long n8 = (long long)N * DDIM / 8;
  for (int it = 0; it < 2; ++it) {
    const float* xc = (it == 0) ? x : (const float*)out;   // iter 2 runs in place (row-wise)
    ghl_f2bf<<<(int)((n8 + 255) / 256), 256, 0, stream>>>(xc, xb, n8);
    hipMemsetAsync(agg, 0, (size_t)N * DDIM * 4, stream);
    ghl_scatter<<<(int)((scat_work + 255) / 256), 256, 0, stream>>>(rowi, coli, xc, dis, agg, scat_work);
    ghl_hopfield<<<(ntiles + WPB - 1) / WPB, WPB * 32, 0, stream>>>(
        xb, Mb, Mt, xc, agg, dis, gma, bta, out, N, ntiles);
  }
}
